// RegressionInstancesModule_16045997818589
// MI455X (gfx1250) — matrix operand order repaired and checked by reading, not compiled, not run on hardware
//
#include <hip/hip_runtime.h>

typedef __attribute__((ext_vector_type(16))) _Float16 v16h;
typedef __attribute__((ext_vector_type(8)))  _Float16 v8h;
typedef __attribute__((ext_vector_type(8)))  float    v8f;

#define BB 4
#define NN 16
#define HH 480
#define WW 640
#define HF 120
#define WF 160
#define CC 128
#define NE 13
#define RR 14
#define MM 64          // BB*NN instances
#define PP 196         // RR*RR output pixels per instance
#define KTOT 1152      // CC*9
#define COTOT 192      // 128 conv_s + 64 conv_c1 output channels fused
#define NT 13          // N tiles of 16 (196 -> 208)
#define LSTRIDE 136    // padded LDS row stride in halves (272 B = 68 words -> conflict-free)
#define ZROW 196       // index of the all-zero LDS row for out-of-image taps
#define PIX_PER_OUT 19660800ull   // BB*NN*HH*WW

// ---------------------------------------------------------------------------
// 1) Pack expert conv weights (f32 -> f16) into A layout [lab][co][K'],
//    K' = tap*128 + ci (tap-major), co<128 from w_conv_s, co>=128 from w_conv_c1.
// ---------------------------------------------------------------------------
__global__ void pack_w_k(const float* __restrict__ ws_,
                         const float* __restrict__ wc1,
                         _Float16* __restrict__ pw) {
    int idx = blockIdx.x * 256 + threadIdx.x;   // exact: NE*COTOT*KTOT multiple of 256
    int kp  = idx % KTOT;                       // K' = tap*128 + ci
    int t   = idx / KTOT;
    int co  = t % COTOT;
    int lab = t / COTOT;
    int tap = kp >> 7;          // 0..8
    int ci  = kp & 127;
    int ksrc = ci * 9 + tap;    // original (ci,ky,kx) row-major index
    float v = (co < 128) ? ws_[((size_t)lab * 128 + co) * KTOT + ksrc]
                         : wc1[((size_t)lab * 64 + (co - 128)) * KTOT + ksrc];
    pw[idx] = (_Float16)v;
}

// ---------------------------------------------------------------------------
// 2) ROI-align: input_feature_map (B,128,120,160) -> roiT f16 (M, P=196, C=128)
//    stored PIXEL-MAJOR so the GEMM's LDS tile is a straight copy.
// ---------------------------------------------------------------------------
__global__ void roi_align_k(const float* __restrict__ fmap,
                            const float* __restrict__ boxes,
                            _Float16* __restrict__ roiT) {
    int m = blockIdx.x;
    int b = m / NN;
    float bf0 = boxes[m * 4 + 0], bf1 = boxes[m * 4 + 1];
    float bf2 = boxes[m * 4 + 2], bf3 = boxes[m * 4 + 3];
    float y1 = 0.6f * bf0, x1 = 0.6f * bf1;
    float y2 = y1 + 0.1f + 0.3f * bf2;
    float x2 = x1 + 0.1f + 0.3f * bf3;
    for (int idx = threadIdx.x; idx < CC * PP; idx += 256) {
        int c = idx / PP, p = idx % PP;
        int i = p / RR, j = p % RR;
        float ty = (i + 0.5f) * (1.0f / RR);
        float tx = (j + 0.5f) * (1.0f / RR);
        float ys = (y1 + (y2 - y1) * ty) * (float)(HF - 1);
        float xs = (x1 + (x2 - x1) * tx) * (float)(WF - 1);
        int y0 = (int)floorf(ys); y0 = min(max(y0, 0), HF - 2);
        int x0 = (int)floorf(xs); x0 = min(max(x0, 0), WF - 2);
        float wy = fminf(fmaxf(ys - (float)y0, 0.f), 1.f);
        float wx = fminf(fmaxf(xs - (float)x0, 0.f), 1.f);
        const float* fp = fmap + ((size_t)(b * CC + c) * HF + y0) * WF + x0;
        float v00 = fp[0], v01 = fp[1], v10 = fp[WF], v11 = fp[WF + 1];
        float v = v00 * (1.f - wy) * (1.f - wx) + v01 * (1.f - wy) * wx
                + v10 * wy * (1.f - wx)        + v11 * wy * wx;
        roiT[(size_t)m * PP * CC + (size_t)p * CC + c] = (_Float16)v;  // pixel-major
    }
}

// ---------------------------------------------------------------------------
// 3) Fused expert GEMM via WMMA, tap-decomposed 3x3 conv:
//    D(192x208) = sum_{tap=0..8} W_tap(192x128) x roiShift_tap(128x208).
//    12 waves, one 16-row M-tile each; LDS tile read-only -> no loop barriers.
//    B fragment = contiguous 32 B slice of padded pixel-major LDS tile;
//    out-of-image columns read the dedicated zero row.
// ---------------------------------------------------------------------------
__global__ __launch_bounds__(384) void expert_gemm_k(
    const _Float16* __restrict__ pw, const _Float16* __restrict__ roiT,
    const float* __restrict__ bs, const float* __restrict__ bc1,
    const int* __restrict__ labels,
    float* __restrict__ gap, _Float16* __restrict__ h1) {
    __shared__ _Float16 sRoi[(PP + 1) * LSTRIDE];   // 197*136*2 = 53584 B

    int m    = blockIdx.x;
    int lab  = labels[m];
    int tid  = threadIdx.x;
    int wid  = tid >> 5;        // 0..11 == M tile
    int lane = tid & 31;
    int hi   = lane >> 4;       // half-wave select
    int n    = lane & 15;       // B column within tile / A row within tile

    // stage ROI tile into padded pixel-major LDS (row p: 128 halves -> stride 136)
    {
        const v8h* src = (const v8h*)(roiT + (size_t)m * PP * CC);
        for (int i = tid; i < PP * (CC / 8); i += 384) {
            int p = i >> 4, c8 = i & 15;
            *(v8h*)&sRoi[p * LSTRIDE + c8 * 8] = src[i];
        }
        if (tid < LSTRIDE / 8) {            // zero row for out-of-image taps
            v8h z;
            #pragma unroll
            for (int e = 0; e < 8; ++e) z[e] = (_Float16)0.f;
            *(v8h*)&sRoi[ZROW * LSTRIDE + tid * 8] = z;
        }
    }
    __syncthreads();

    v8f acc[NT];
    #pragma unroll
    for (int t = 0; t < NT; ++t)
        #pragma unroll
        for (int e = 0; e < 8; ++e) acc[t][e] = 0.f;

    const _Float16* arow = pw + ((size_t)lab * COTOT + wid * 16 + n) * KTOT;

    for (int tap = 0; tap < 9; ++tap) {
        int dy = tap / 3 - 1, dx = tap % 3 - 1;
        // per-lane shifted-pixel row offsets (zero row when outside the image)
        int poff[NT];
        #pragma unroll
        for (int t = 0; t < NT; ++t) {
            int p = t * 16 + n;
            int y = p / RR + dy;
            int x = p % RR + dx;
            bool v = (p < PP) && ((unsigned)y < RR) && ((unsigned)x < RR);
            poff[t] = (v ? (y * RR + x) : ZROW) * LSTRIDE;
        }
        #pragma unroll
        for (int kc = 0; kc < 4; ++kc) {
            int kp = tap * 128 + kc * 32;
            // A fragment (16x32 f16): two contiguous 16 B global chunks per lane
            v16h a;
            {
                const v8h* a0 = (const v8h*)(arow + kp + hi * 16);
                const v8h* a1 = (const v8h*)(arow + kp + hi * 16 + 8);
                v8h lo = *a0, hh = *a1;
                #pragma unroll
                for (int e = 0; e < 8; ++e) { a[e] = lo[e]; a[8 + e] = hh[e]; }
            }
            // B fragments: lane holds channels kc*32 + hi*16 .. +15 at its column
            #pragma unroll
            for (int t = 0; t < NT; ++t) {
                const v8h* bp = (const v8h*)&sRoi[poff[t] + kc * 32 + hi * 16];
                v8h blo = bp[0], bhh = bp[1];
                v16h bfr;
                #pragma unroll
                for (int e = 0; e < 8; ++e) { bfr[e] = blo[e]; bfr[8 + e] = bhh[e]; }
                acc[t] = __builtin_amdgcn_wmma_f32_16x16x32_f16(
                    false, a, false, bfr, (short)0, acc[t], false, false);
            }
        }
    }

    // epilogue: D element (r, lane) -> channel co = wid*16 + r + 8*hi, pixel p = t*16 + n
    if (wid < 8) {  // conv_s rows -> bias + relu + global average pool
        float bias[8], gsum[8];
        #pragma unroll
        for (int r = 0; r < 8; ++r) {
            bias[r] = bs[lab * CC + wid * 16 + r + 8 * hi];
            gsum[r] = 0.f;
        }
        #pragma unroll
        for (int t = 0; t < NT; ++t) {
            int p = t * 16 + n;
            if (p < PP) {
                #pragma unroll
                for (int r = 0; r < 8; ++r)
                    gsum[r] += fmaxf(acc[t][r] + bias[r], 0.f);
            }
        }
        #pragma unroll
        for (int r = 0; r < 8; ++r) {   // butterfly over the 16-lane half (bits 0..3)
            gsum[r] += __shfl_xor(gsum[r], 1, 32);
            gsum[r] += __shfl_xor(gsum[r], 2, 32);
            gsum[r] += __shfl_xor(gsum[r], 4, 32);
            gsum[r] += __shfl_xor(gsum[r], 8, 32);
        }
        if (n == 0) {
            #pragma unroll
            for (int r = 0; r < 8; ++r)
                gap[m * CC + wid * 16 + r + 8 * hi] = gsum[r] * (1.f / PP);
        }
    } else {        // conv_c1 rows -> bias + relu -> h1 (f16)
        float bias[8];
        #pragma unroll
        for (int r = 0; r < 8; ++r)
            bias[r] = bc1[lab * 64 + (wid - 8) * 16 + r + 8 * hi];
        #pragma unroll
        for (int t = 0; t < NT; ++t) {
            int p = t * 16 + n;
            if (p < PP) {
                #pragma unroll
                for (int r = 0; r < 8; ++r) {
                    int co2 = (wid - 8) * 16 + r + 8 * hi;
                    h1[((size_t)m * 64 + co2) * PP + p] =
                        (_Float16)fmaxf(acc[t][r] + bias[r], 0.f);
                }
            }
        }
    }
}

// ---------------------------------------------------------------------------
// 4) FC: scale/shift = gap(128) . w_fc[lab](2,128) + b_fc[lab]
// ---------------------------------------------------------------------------
__global__ void fc_k(const float* __restrict__ gap, const float* __restrict__ wfc,
                     const float* __restrict__ bfc, const int* __restrict__ labels,
                     float* __restrict__ scale, float* __restrict__ shift,
                     float* __restrict__ outTail) {
    int m = blockIdx.x, lane = threadIdx.x;
    int lab = labels[m];
    float d0 = 0.f, d1 = 0.f;
    for (int c = lane; c < CC; c += 32) {
        float g = gap[m * CC + c];
        d0 += g * wfc[(lab * 2 + 0) * CC + c];
        d1 += g * wfc[(lab * 2 + 1) * CC + c];
    }
    for (int o = 16; o > 0; o >>= 1) {
        d0 += __shfl_xor(d0, o, 32);
        d1 += __shfl_xor(d1, o, 32);
    }
    if (lane == 0) {
        float sc = d0 + bfc[lab * 2 + 0];
        float sh = d1 + bfc[lab * 2 + 1];
        scale[m] = sc; shift[m] = sh;
        outTail[m] = sc; outTail[MM + m] = sh;
    }
}

// ---------------------------------------------------------------------------
// 5) conv_c2: (64,14,14) -> (1,14,14), tiny; h1 + weights staged in LDS
// ---------------------------------------------------------------------------
__global__ void conv_c2_k(const _Float16* __restrict__ h1, const float* __restrict__ w2,
                          const float* __restrict__ b2, const int* __restrict__ labels,
                          float* __restrict__ cout) {
    __shared__ float    sW[64 * 9];
    __shared__ _Float16 sH[64 * PP];
    int m = blockIdx.x, tid = threadIdx.x, lab = labels[m];
    for (int i = tid; i < 64 * 9; i += 256) sW[i] = w2[lab * 64 * 9 + i];
    {
        const v8h* src = (const v8h*)(h1 + (size_t)m * 64 * PP);
        v8h* dst = (v8h*)sH;
        for (int i = tid; i < (64 * PP) / 8; i += 256) dst[i] = src[i];
    }
    __syncthreads();
    if (tid < PP) {
        int y = tid / RR, x = tid % RR;
        float acc = b2[lab];
        for (int ci = 0; ci < 64; ++ci) {
            const _Float16* hp = sH + ci * PP;
            const float* wp = sW + ci * 9;
            #pragma unroll
            for (int dy = 0; dy < 3; ++dy) {
                int yy = y + dy - 1;
                if (yy < 0 || yy >= RR) continue;
                #pragma unroll
                for (int dx = 0; dx < 3; ++dx) {
                    int xx = x + dx - 1;
                    if (xx < 0 || xx >= RR) continue;
                    acc += (float)hp[yy * RR + xx] * wp[dy * 3 + dx];
                }
            }
        }
        cout[m * PP + tid] = acc;
    }
}

// ---------------------------------------------------------------------------
// 6) Bilinear upsample (half-pixel, edge clamp) 14x14 -> 480x640 + affine clamp.
//    157 MB of stores -> the HBM-bound stage; float4 writes, tile in LDS.
// ---------------------------------------------------------------------------
__global__ void resize_final_k(const float* __restrict__ cmap,
                               const float* __restrict__ scale,
                               const float* __restrict__ shift,
                               float* __restrict__ depth_out,
                               float* __restrict__ canon_out) {
    __shared__ float sC[PP];
    int m = blockIdx.y, tid = threadIdx.x;
    if (tid < PP) sC[tid] = cmap[m * PP + tid];
    __syncthreads();
    float sc = scale[m], sh = shift[m];
    int pix4 = (blockIdx.x * 256 + tid) * 4;
    int y  = pix4 / WW;
    int x0 = pix4 - y * WW;
    float fy = ((float)y + 0.5f) * (14.f / 480.f) - 0.5f;
    int   iy = (int)floorf(fy);
    float wy = fy - (float)iy;
    int ya = min(max(iy, 0), RR - 1), yb = min(max(iy + 1, 0), RR - 1);
    float4 cv, dv;
    float* cvp = &cv.x; float* dvp = &dv.x;
    #pragma unroll
    for (int j = 0; j < 4; ++j) {
        float fx = ((float)(x0 + j) + 0.5f) * (14.f / 640.f) - 0.5f;
        int   ix = (int)floorf(fx);
        float wx = fx - (float)ix;
        int xa = min(max(ix, 0), RR - 1), xb = min(max(ix + 1, 0), RR - 1);
        float v = (sC[ya * RR + xa] * (1.f - wx) + sC[ya * RR + xb] * wx) * (1.f - wy)
                + (sC[yb * RR + xa] * (1.f - wx) + sC[yb * RR + xb] * wx) * wy;
        cvp[j] = v;
        dvp[j] = fmaxf(v * sc + sh, 0.001f);
    }
    size_t base = (size_t)m * HH * WW + (size_t)pix4;
    *(float4*)(canon_out + base) = cv;
    *(float4*)(depth_out + base) = dv;
}

// ---------------------------------------------------------------------------
extern "C" void kernel_launch(void* const* d_in, const int* in_sizes, int n_in,
                              void* d_out, int out_size, void* d_ws, size_t ws_size,
                              hipStream_t stream) {
    const float* fmap      = (const float*)d_in[2];
    const float* boxes     = (const float*)d_in[8];
    const int*   labels    = (const int*)d_in[9];
    const float* w_conv_s  = (const float*)d_in[10];
    const float* b_conv_s  = (const float*)d_in[11];
    const float* w_fc      = (const float*)d_in[12];
    const float* b_fc      = (const float*)d_in[13];
    const float* w_conv_c1 = (const float*)d_in[14];
    const float* b_conv_c1 = (const float*)d_in[15];
    const float* w_conv_c2 = (const float*)d_in[16];
    const float* b_conv_c2 = (const float*)d_in[17];
    float* out = (float*)d_out;

    char* ws = (char*)d_ws;
    size_t off = 0;
    auto alloc = [&](size_t bytes) -> void* {
        void* p = ws + off;
        off = (off + bytes + 255) & ~(size_t)255;
        return p;
    };
    _Float16* pw   = (_Float16*)alloc((size_t)NE * COTOT * KTOT * 2);  // 5.75 MB
    _Float16* roiT = (_Float16*)alloc((size_t)MM * PP * CC * 2);       // 3.2 MB
    float*    gap  = (float*)   alloc((size_t)MM * CC * 4);
    _Float16* h1   = (_Float16*)alloc((size_t)MM * 64 * PP * 2);       // 1.6 MB
    float*    cmap = (float*)   alloc((size_t)MM * PP * 4);
    float*    scl  = (float*)   alloc((size_t)MM * 4);
    float*    shf  = (float*)   alloc((size_t)MM * 4);

    pack_w_k<<<(NE * COTOT * KTOT) / 256, 256, 0, stream>>>(w_conv_s, w_conv_c1, pw);
    roi_align_k<<<MM, 256, 0, stream>>>(fmap, boxes, roiT);
    expert_gemm_k<<<MM, 384, 0, stream>>>(pw, roiT, b_conv_s, b_conv_c1, labels, gap, h1);
    fc_k<<<MM, 32, 0, stream>>>(gap, w_fc, b_fc, labels, scl, shf,
                                out + 2ull * PIX_PER_OUT);
    conv_c2_k<<<MM, 256, 0, stream>>>(h1, w_conv_c2, b_conv_c2, labels, cmap);
    resize_final_k<<<dim3(300, MM), 256, 0, stream>>>(cmap, scl, shf,
                                                      out, out + PIX_PER_OUT);
}
